// Conv2d_NN_7559142441290
// MI455X (gfx1250) — compile-verified, matrix operand order checked by reading, not run on hardware
//
#include <hip/hip_runtime.h>
#include <hip/hip_bf16.h>
#include <stddef.h>

// ---------------- problem constants ----------------
#define BATCH   32
#define C_IN    32
#define C_OUT   64
#define HH      96
#define WW      96
#define KNN     9
#define N_TOK   2304          // (96/2)*(96/2)
#define D_FEAT  128           // 32*2*2
#define D_OUT   256           // 64*2*2
#define K_GEMM  1152          // 128*9
#define CHUNK   576           // column chunk for distance buffering
#define NCHUNK  4             // 4*576 = 2304

// workspace layout (bytes)
#define XT_BYTES   ((size_t)BATCH * N_TOK * D_FEAT * 4)        // 37,748,736 (f32 tokens)
#define XH_BYTES   ((size_t)BATCH * N_TOK * D_FEAT * 2)        // 18,874,368 (f16 tokens)
#define NSQ_BYTES  ((size_t)BATCH * N_TOK * 4)                 //    294,912
#define WH_BYTES   ((size_t)D_OUT * K_GEMM * 2)                //    589,824
#define OFF_XT   ((size_t)0)
#define OFF_XH   (OFF_XT + XT_BYTES)
#define OFF_NSQ  (OFF_XH + XH_BYTES)
#define OFF_WH   (OFF_NSQ + NSQ_BYTES)
#define OFF_IDX  (OFF_WH + WH_BYTES)

typedef __attribute__((ext_vector_type(2)))  float     v2f;
typedef __attribute__((ext_vector_type(8)))  float     v8f;
typedef __attribute__((ext_vector_type(8)))  _Float16  v8h;
typedef __attribute__((ext_vector_type(16))) _Float16  v16h;

// ---------------------------------------------------------------------------
// K0: pixel_unshuffle + transpose -> Xt (f32) and Xh (f16), plus nsq norms
// ---------------------------------------------------------------------------
__global__ __launch_bounds__(256) void prep_tokens_kernel(
    const float* __restrict__ x, float* __restrict__ Xt,
    _Float16* __restrict__ Xh, float* __restrict__ nsq) {
  int tok = blockIdx.x * blockDim.x + threadIdx.x;
  if (tok >= BATCH * N_TOK) return;
  int b = tok / N_TOK;
  int n = tok - b * N_TOK;
  int hh = n / 48, ww = n - hh * 48;
  float ss = 0.0f;
  float*    dst  = Xt + (size_t)tok * D_FEAT;
  _Float16* dsth = Xh + (size_t)tok * D_FEAT;
  #pragma unroll 4
  for (int d = 0; d < D_FEAT; ++d) {
    int c  = d >> 2;
    int sy = (d >> 1) & 1;
    int sx = d & 1;
    float v = x[(((size_t)b * C_IN + c) * HH + (hh * 2 + sy)) * WW + (ww * 2 + sx)];
    dst[d]  = v;
    dsth[d] = (_Float16)v;
    ss += v * v;
  }
  nsq[tok] = ss;
}

// ---------------------------------------------------------------------------
// KW: reorder conv_w (o, i, k) -> Wh[o][k*128+i] as f16
// ---------------------------------------------------------------------------
__global__ __launch_bounds__(256) void prep_weights_kernel(
    const float* __restrict__ conv_w, _Float16* __restrict__ Wh) {
  int g = blockIdx.x * blockDim.x + threadIdx.x;
  if (g >= D_OUT * K_GEMM) return;
  int o = g / K_GEMM;
  int col = g - o * K_GEMM;
  int k = col >> 7;        // col / 128
  int i = col & 127;       // col % 128
  Wh[g] = (_Float16)conv_w[(size_t)o * K_GEMM + i * KNN + k];
}

// sorted-ascending (dist, idx) insertion, tie-break = lower index (matches top_k)
__device__ __forceinline__ void topk_insert(float (&bd)[KNN], int (&bi)[KNN],
                                            float d, int j) {
  if (d < bd[KNN - 1] || (d == bd[KNN - 1] && j < bi[KNN - 1])) {
    bd[KNN - 1] = d; bi[KNN - 1] = j;
    #pragma unroll
    for (int q = KNN - 1; q >= 1; --q) {
      bool sw = (bd[q] < bd[q - 1]) || ((bd[q] == bd[q - 1]) && (bi[q] < bi[q - 1]));
      if (sw) {
        float td = bd[q]; bd[q] = bd[q - 1]; bd[q - 1] = td;
        int   ti = bi[q]; bi[q] = bi[q - 1]; bi[q - 1] = ti;
      }
    }
  }
}

// ---------------------------------------------------------------------------
// K1: f32-WMMA pairwise scores + fused per-row top-9
// A row-block staged once in LDS (shared by all 4 waves, conflict-free pad),
// B streamed from global/L2.
// grid: (144 row-tiles, 32 batches), block: 128 (4 waves)
// ---------------------------------------------------------------------------
#define A_LD (D_FEAT + 4)    // pad 4 floats: lane m -> banks 4m..4m+3, all 64 used
__global__ __launch_bounds__(128) void knn_kernel(
    const float* __restrict__ Xt, const float* __restrict__ nsq,
    int* __restrict__ idxOut) {
  __shared__ __align__(8) float rowsA[16][A_LD];   // 8,448 B
  __shared__ float distBuf[16][CHUNK + 1];         // 36,928 B (+1 pad: bank stagger)
  __shared__ float candD[16][72];
  __shared__ int   candI[16][72];

  const int rt   = blockIdx.x;               // row tile 0..143
  const int b    = blockIdx.y;
  const int tid  = threadIdx.x;
  const int wave = tid >> 5;
  const int lane = tid & 31;
  const int lmod = lane & 15;
  const int khalf = (lane < 16) ? 0 : 2;     // f32 A/B: K pair select per half-wave
  const int rOff  = (lane < 16) ? 0 : 8;     // C layout: rows v / v+8

  const float* XtB = Xt + (size_t)b * N_TOK * D_FEAT;

  // cooperative load of the A row-block (16 rows x 128 feats), reused 144x
  for (int e = tid; e < 16 * D_FEAT; e += 128) {
    int r = e >> 7, c = e & 127;
    rowsA[r][c] = XtB[(size_t)(rt * 16 + r) * D_FEAT + c];
  }
  __syncthreads();

  // running top-9 per (row = tid>>3, sub = tid&7)
  float bd[KNN]; int bi[KNN];
  #pragma unroll
  for (int q = 0; q < KNN; ++q) { bd[q] = 3.0e38f; bi[q] = 0x7fffffff; }
  const int selRow = tid >> 3;
  const int selSub = tid & 7;

  for (int ch = 0; ch < NCHUNK; ++ch) {
    // ---- WMMA compute: 36 column tiles per chunk, 9 per wave ----
    for (int t = wave; t < CHUNK / 16; t += 4) {
      const int cb = ch * CHUNK + t * 16;
      const float* colBase = XtB + (size_t)(cb + lmod) * D_FEAT;
      v8f acc = {};
      #pragma unroll 8
      for (int k0 = 0; k0 < D_FEAT; k0 += 4) {
        v2f av = *(const v2f*)&rowsA[lmod][k0 + khalf];
        v2f bv = *(const v2f*)(colBase + k0 + khalf);
        acc = __builtin_amdgcn_wmma_f32_16x16x4_f32(
            false, av, false, bv, (short)0, acc, false, false);
      }
      const float ns = nsq[b * N_TOK + cb + lmod];
      #pragma unroll
      for (int v = 0; v < 8; ++v) {
        // per-row-constant nsq[row] dropped (monotone for top-k)
        distBuf[v + rOff][t * 16 + lmod] = ns - 2.0f * acc[v];
      }
    }
    __syncthreads();
    // ---- selection: 8 threads per row scan this chunk ----
    for (int c = selSub; c < CHUNK; c += 8) {
      topk_insert(bd, bi, distBuf[selRow][c], ch * CHUNK + c);
    }
    __syncthreads();
  }

  // ---- 72 -> 9 merge per row ----
  #pragma unroll
  for (int q = 0; q < KNN; ++q) {
    candD[selRow][selSub * KNN + q] = bd[q];
    candI[selRow][selSub * KNN + q] = bi[q];
  }
  __syncthreads();
  if (tid < 16) {
    float fd[KNN]; int fi[KNN];
    #pragma unroll
    for (int q = 0; q < KNN; ++q) { fd[q] = 3.0e38f; fi[q] = 0x7fffffff; }
    for (int m = 0; m < 72; ++m) topk_insert(fd, fi, candD[tid][m], candI[tid][m]);
    const int rowG = rt * 16 + tid;
    #pragma unroll
    for (int q = 0; q < KNN; ++q)
      idxOut[((size_t)b * N_TOK + rowG) * KNN + q] = fi[q];
  }
}

// ---------------------------------------------------------------------------
// K2: async-DMA gather of neighbor rows (f16) -> LDS, f16-WMMA GEMM
//     (256 x 1152 x 16), bias + ReLU + fused pixel_shuffle store
// grid: (144 token-tiles, 32 batches), block: 128 (4 waves, 4 M-tiles each)
// ---------------------------------------------------------------------------
#define GT_LD (K_GEMM + 8)   // pad 8 halves (16B) to stagger banks
__global__ __launch_bounds__(128) void conv_kernel(
    const _Float16* __restrict__ Xh, const int* __restrict__ idxIn,
    const _Float16* __restrict__ Wh, const float* __restrict__ bias,
    float* __restrict__ out) {
  __shared__ __align__(16) _Float16 gt[16][GT_LD];

  const int tt   = blockIdx.x;               // token tile 0..143
  const int b    = blockIdx.y;
  const int tid  = threadIdx.x;
  const int wave = tid >> 5;
  const int lane = tid & 31;
  const int lmod = lane & 15;
  const int tokBase = tt * 16;
  const _Float16* XhB = Xh + (size_t)b * N_TOK * D_FEAT;

  // ---- gather: async DMA 16-byte segments (8 halves) straight into LDS ----
  // element space: 16 tokens x 9 neighbors x 16 segments = 2304; 18 per thread
  for (int e = tid; e < 16 * KNN * 16; e += 128) {
    int t   = e / (KNN * 16);
    int r   = e - t * (KNN * 16);
    int k   = r >> 4;          // neighbor 0..8
    int seg = r & 15;          // 8-half segment 0..15
    int j = idxIn[((size_t)b * N_TOK + tokBase + t) * KNN + k];
    unsigned long long gaddr =
        (unsigned long long)(const void*)(XhB + (size_t)j * D_FEAT + seg * 8);
    // low 32 bits of a generic __shared__ address == LDS byte offset
    unsigned ldsOff = (unsigned)(size_t)(const void*)&gt[t][k * 128 + seg * 8];
    asm volatile("global_load_async_to_lds_b128 %0, %1, off"
                 :: "v"(ldsOff), "v"(gaddr) : "memory");
  }
  asm volatile("s_wait_asynccnt 0x0" ::: "memory");
  __syncthreads();

  const int hoff = (lane < 16) ? 0 : 8;      // f16 A/B: K lo/hi 8 per half-wave
  const int rOff = (lane < 16) ? 0 : 8;
  const int n  = tokBase + lmod;
  const int hh = n / 48, ww2 = n - (n / 48) * 48;

  #pragma unroll
  for (int mtl = 0; mtl < 4; ++mtl) {
    const int mt = wave * 4 + mtl;
    const _Float16* wrow = Wh + (size_t)(mt * 16 + lmod) * K_GEMM;
    v8f acc = {};
    for (int k0 = 0; k0 < K_GEMM; k0 += 32) {
      v8h alo = *(const v8h*)(wrow + k0 + hoff);
      v8h ahi = *(const v8h*)(wrow + k0 + 16 + hoff);
      v8h blo = *(const v8h*)(&gt[lmod][k0 + hoff]);
      v8h bhi = *(const v8h*)(&gt[lmod][k0 + 16 + hoff]);
      v16h av = __builtin_shufflevector(alo, ahi, 0,1,2,3,4,5,6,7,8,9,10,11,12,13,14,15);
      v16h bv = __builtin_shufflevector(blo, bhi, 0,1,2,3,4,5,6,7,8,9,10,11,12,13,14,15);
      acc = __builtin_amdgcn_wmma_f32_16x16x32_f16(
          false, av, false, bv, (short)0, acc, false, false);
    }
    #pragma unroll
    for (int v = 0; v < 8; ++v) {
      int o = mt * 16 + v + rOff;
      float val = acc[v] + bias[o];
      val = val > 0.0f ? val : 0.0f;
      int co = o >> 2, sy = (o >> 1) & 1, sx = o & 1;
      size_t oi = (((size_t)b * C_OUT + co) * HH + (hh * 2 + sy)) * WW + (ww2 * 2 + sx);
      out[oi] = val;
    }
  }
}

// ---------------------------------------------------------------------------
extern "C" void kernel_launch(void* const* d_in, const int* in_sizes, int n_in,
                              void* d_out, int out_size, void* d_ws, size_t ws_size,
                              hipStream_t stream) {
  const float* x      = (const float*)d_in[0];   // (32,32,96,96)
  const float* conv_w = (const float*)d_in[1];   // (256,128,9)
  const float* conv_b = (const float*)d_in[2];   // (256,)
  float* out = (float*)d_out;                    // (32,64,96,96)

  char* ws = (char*)d_ws;
  float*    Xt  = (float*)(ws + OFF_XT);
  _Float16* Xh  = (_Float16*)(ws + OFF_XH);
  float*    nsq = (float*)(ws + OFF_NSQ);
  _Float16* Wh  = (_Float16*)(ws + OFF_WH);
  int*      idx = (int*)(ws + OFF_IDX);

  // K0: tokens (f32 + f16) + norms
  {
    int total = BATCH * N_TOK;
    prep_tokens_kernel<<<(total + 255) / 256, 256, 0, stream>>>(x, Xt, Xh, nsq);
  }
  // KW: weight reorder -> f16
  {
    int total = D_OUT * K_GEMM;
    prep_weights_kernel<<<(total + 255) / 256, 256, 0, stream>>>(conv_w, Wh);
  }
  // K1: f32-WMMA distances + fused top-9
  knn_kernel<<<dim3(N_TOK / 16, BATCH), 128, 0, stream>>>(Xt, nsq, idx);
  // K2: async gather + f16-WMMA conv GEMM + epilogue
  conv_kernel<<<dim3(N_TOK / 16, BATCH), 128, 0, stream>>>(Xh, idx, Wh, conv_b, out);
}
